// TransitionModel_90512140796462
// MI455X (gfx1250) — matrix operand order repaired and checked by reading, not compiled, not run on hardware
//
#include <hip/hip_runtime.h>

typedef __attribute__((ext_vector_type(16))) _Float16 v16h;
typedef __attribute__((ext_vector_type(8)))  _Float16 v8h;
typedef __attribute__((ext_vector_type(8)))  float    v8f;

#define T_LEN   131072
#define KDIM    256
#define NCHUNK  256
#define CLEN    (T_LEN / NCHUNK)    // 512
#define PSTR    264                 // padded LDS row stride in halfs (16B-aligned pad)

// workspace layout (bytes, all 256-aligned)
#define WS_TS16 ((size_t)0)         // 2*256*256 f16  = 262144
#define WS_ACT  ((size_t)262144)    // T bytes        = 131072
#define WS_S0   ((size_t)393216)    // 256 f32 (pad to 1024)
#define WS_CST  ((size_t)394240)    // NCHUNK*256 f32 = 262144
#define WS_MT   ((size_t)656384)    // NCHUNK*256*256 f16 = 33554432

// ---------------------------------------------------------------------------
// Column-wise softmax of transition[2,K,K] over axis=1 (rows), emit f16 copy.
__global__ void softmax_T_kernel(const float* __restrict__ trans,
                                 _Float16* __restrict__ ts16) {
  __shared__ float red[256];
  int b = blockIdx.x;               // 0..511 : (action, column)
  int a = b >> 8, j = b & 255;
  int i = threadIdx.x;
  float v = trans[((size_t)a << 16) + (size_t)i * 256 + j];
  red[i] = v; __syncthreads();
  for (int s = 128; s > 0; s >>= 1) {
    if (i < s) red[i] = fmaxf(red[i], red[i + s]);
    __syncthreads();
  }
  float mx = red[0]; __syncthreads();
  float e = expf(v - mx);
  red[i] = e; __syncthreads();
  for (int s = 128; s > 0; s >>= 1) {
    if (i < s) red[i] += red[i + s];
    __syncthreads();
  }
  float p = e / red[0];
  ts16[((size_t)a << 16) + (size_t)i * 256 + j] = (_Float16)p;
}

// ---------------------------------------------------------------------------
// actions[t] = 0 if int(traj[t,5]) == -1 else 1   (trunc-toward-zero cast)
__global__ void actions_kernel(const float* __restrict__ traj,
                               unsigned char* __restrict__ act) {
  int t = blockIdx.x * 256 + threadIdx.x;
  float v = traj[(size_t)t * 6 + 5];
  int ai = (int)v;
  act[t] = (ai == -1) ? 0 : 1;
}

// ---------------------------------------------------------------------------
// s0 = softmax( enc_w2 @ relu(enc_w1 @ traj[0,:2] + enc_b1) + enc_b2 )
__global__ void s0_kernel(const float* __restrict__ traj,
                          const float* __restrict__ w1, const float* __restrict__ b1,
                          const float* __restrict__ w2, const float* __restrict__ b2,
                          float* __restrict__ s0) {
  __shared__ float h[16];
  __shared__ float red[256];
  int k = threadIdx.x;
  if (k < 16) {
    float hv = traj[0] * w1[k * 2] + traj[1] * w1[k * 2 + 1] + b1[k];
    h[k] = fmaxf(hv, 0.f);
  }
  __syncthreads();
  float lg = b2[k];
  for (int i = 0; i < 16; ++i) lg += w2[k * 16 + i] * h[i];
  red[k] = lg; __syncthreads();
  for (int s = 128; s > 0; s >>= 1) {
    if (k < s) red[k] = fmaxf(red[k], red[k + s]);
    __syncthreads();
  }
  float mx = red[0]; __syncthreads();
  float e = expf(lg - mx);
  red[k] = e; __syncthreads();
  for (int s = 128; s > 0; s >>= 1) {
    if (k < s) red[k] += red[k + s];
    __syncthreads();
  }
  s0[k] = e / red[0];
}

// ---------------------------------------------------------------------------
// Per-chunk matrix product M_c = T_{a_{t0+L-1}} ... T_{a_{t0}} via WMMA.
// Running product P kept TRANSPOSED in LDS as f16 (PT[n][k] = P[k][n]) so
// B-fragments (2 runs of 8 contiguous K per lane) and D-writebacks (8 contiguous M per
// lane) are aligned 16B LDS accesses.  Output: mt_out[c][n*256+k] = M_c[k][n].
__global__ void __launch_bounds__(1024)
chunk_prod_kernel(const _Float16* __restrict__ ts16,
                  const unsigned char* __restrict__ act,
                  _Float16* __restrict__ mt_out) {
  extern __shared__ char smem[];
  _Float16* PT = (_Float16*)smem;   // [256][PSTR] f16 = 132KB
  const int c    = blockIdx.x;
  const int tid  = threadIdx.x;
  const int lane = tid & 31;
  const int wv   = tid >> 5;        // 32 waves
  const int r    = lane & 15;
  const int hs   = lane >> 4;
  // wave tiling: 2 M-tiles x 4 N-tiles of 16x16 each -> 8 f32 accum tiles/wave
  const int mtb = (wv >> 2) * 2;    // 0,2,..,14
  const int ntb = (wv & 3) * 4;     // 0,4,8,12

  const int t0 = c * CLEN;
  // PT = transpose(T_{a_{t0}})
  {
    const _Float16* Tm0 = ts16 + ((size_t)act[t0] << 16);
    for (int idx = tid; idx < 65536; idx += 1024) {
      int k = idx >> 8, n = idx & 255;
      PT[n * PSTR + k] = Tm0[(size_t)k * 256 + n];
    }
  }
  __syncthreads();

  v8f acc[2][4];
  for (int t = t0 + 1; t < t0 + CLEN; ++t) {
    const _Float16* Ta = ts16 + ((size_t)act[t] << 16);
    for (int mi = 0; mi < 2; ++mi)
      for (int ni = 0; ni < 4; ++ni)
        for (int e = 0; e < 8; ++e) acc[mi][ni][e] = 0.f;

    for (int kt = 0; kt < 8; ++kt) {
      // A fragments: row = mtile*16 + r ; K = kt*32 + 8*hs + {0..7} and +16
      v16h afrag[2];
      for (int mi = 0; mi < 2; ++mi) {
        const _Float16* p0 = Ta + (size_t)((mtb + mi) * 16 + r) * 256 + kt * 32 + 8 * hs;
        v8h lo = *(const v8h*)p0;
        v8h hi = *(const v8h*)(p0 + 16);
        v16h a;
        for (int e = 0; e < 8; ++e) { a[e] = lo[e]; a[e + 8] = hi[e]; }
        afrag[mi] = a;
      }
      for (int ni = 0; ni < 4; ++ni) {
        // B fragment: col n = ntile*16 + r ; K = kt*32 + 8*hs + {0..7} and +16
        const _Float16* bp = PT + ((ntb + ni) * 16 + r) * PSTR + kt * 32 + 8 * hs;
        v8h b0 = *(const v8h*)bp;
        v8h b1 = *(const v8h*)(bp + 16);
        v16h b;
        for (int e = 0; e < 8; ++e) { b[e] = b0[e]; b[e + 8] = b1[e]; }
        for (int mi = 0; mi < 2; ++mi) {
          acc[mi][ni] = __builtin_amdgcn_wmma_f32_16x16x32_f16(
              false, afrag[mi], false, b, (short)0, acc[mi][ni], false, false);
        }
      }
    }
    __syncthreads();            // all waves done reading old PT
    // D tile: lane col n = ntile*16 + r ; rows M = mtile*16 + 8*hs + e (contig)
    for (int mi = 0; mi < 2; ++mi)
      for (int ni = 0; ni < 4; ++ni) {
        int n  = (ntb + ni) * 16 + r;
        int m0 = (mtb + mi) * 16 + 8 * hs;
        v8h hv;
        for (int e = 0; e < 8; ++e) hv[e] = (_Float16)acc[mi][ni][e];
        *(v8h*)(PT + n * PSTR + m0) = hv;
      }
    __syncthreads();
  }

  // dump transposed product to global (contiguous)
  _Float16* out = mt_out + ((size_t)c << 16);
  for (int idx = tid; idx < 65536; idx += 1024) {
    int n = idx >> 8, k = idx & 255;
    out[(size_t)n * 256 + k] = PT[n * PSTR + k];
  }
}

// ---------------------------------------------------------------------------
// Sequential scan over chunk matrices -> chunk start states.
// mt[c][n*256+m] = M_c[m][n]  =>  s'[m] = sum_n mt[c][n*256+m] * s[n]  (coalesced)
__global__ void scan_kernel(const _Float16* __restrict__ mt,
                            const float* __restrict__ s0,
                            float* __restrict__ cst) {
  __shared__ float s[256];
  int i = threadIdx.x;
  s[i] = s0[i];
  cst[i] = s0[i];
  __syncthreads();
  for (int c = 0; c < NCHUNK - 1; ++c) {
    const _Float16* M = mt + ((size_t)c << 16);
    float acc = 0.f;
    for (int n = 0; n < 256; ++n)
      acc += (float)M[(size_t)n * 256 + i] * s[n];
    __syncthreads();
    s[i] = acc;
    cst[(size_t)(c + 1) * 256 + i] = acc;
    __syncthreads();
  }
}

// ---------------------------------------------------------------------------
// Replay matvecs within each chunk; both f16 transition matrices LDS-resident.
__global__ void __launch_bounds__(1024)
replay_kernel(const _Float16* __restrict__ ts16,
              const unsigned char* __restrict__ act,
              const float* __restrict__ cst,
              float* __restrict__ preds) {
  extern __shared__ char smem[];
  _Float16* Tl = (_Float16*)smem;                         // [2][256][PSTR]
  float* sbuf  = (float*)(smem + (size_t)2 * 256 * PSTR * 2);  // 2 x 256 f32
  unsigned char* actl = (unsigned char*)(sbuf + 512);     // CLEN bytes
  const int c = blockIdx.x, tid = threadIdx.x;

  for (int idx = tid; idx < 2 * 65536; idx += 1024) {
    int a = idx >> 16, rem = idx & 65535, i = rem >> 8, j = rem & 255;
    Tl[(size_t)(a * 256 + i) * PSTR + j] = ts16[idx];
  }
  for (int idx = tid; idx < CLEN; idx += 1024) actl[idx] = act[c * CLEN + idx];
  if (tid < 256) sbuf[tid] = cst[(size_t)c * 256 + tid];
  __syncthreads();

  const int i = tid >> 2, q = tid & 3;   // 4 lanes cooperate per output row
  float* scur = sbuf;
  float* snxt = sbuf + 256;
  for (int tt = 0; tt < CLEN; ++tt) {
    int a = actl[tt];
    const _Float16* trow = Tl + (size_t)(a * 256 + i) * PSTR + q * 64;
    const float* sv = scur + q * 64;
    float acc = 0.f;
    for (int j = 0; j < 64; ++j) acc += (float)trow[j] * sv[j];
    acc += __shfl_xor(acc, 1);
    acc += __shfl_xor(acc, 2);
    if (q == 0) {
      snxt[i] = acc;
      preds[((size_t)(c * CLEN + tt)) * 256 + i] = acc;
    }
    __syncthreads();
    float* tmp = scur; scur = snxt; snxt = tmp;
  }
}

// ---------------------------------------------------------------------------
// decoded[t] = dec_w2 @ relu(dec_w1 @ states[t] + dec_b1) + dec_b2
// states[t] = s0 for t==0 else preds[t-1].  One wave per timestep.
__global__ void decode_kernel(const float* __restrict__ preds,
                              const float* __restrict__ s0,
                              const float* __restrict__ w1, const float* __restrict__ b1,
                              const float* __restrict__ w2, const float* __restrict__ b2,
                              float* __restrict__ dec) {
  const int lane = threadIdx.x & 31;
  const int wv = threadIdx.x >> 5;
  const int t = blockIdx.x * 8 + wv;
  const float* s = (t == 0) ? s0 : (preds + ((size_t)(t - 1) << 8));
  float sv[8];
  const int j0 = lane * 8;
  for (int e = 0; e < 8; ++e) sv[e] = s[j0 + e];
  float o0 = b2[0], o1 = b2[1];
  for (int rr = 0; rr < 16; ++rr) {
    float p = 0.f;
    for (int e = 0; e < 8; ++e) p += w1[(size_t)rr * 256 + j0 + e] * sv[e];
    p += __shfl_xor(p, 16);
    p += __shfl_xor(p, 8);
    p += __shfl_xor(p, 4);
    p += __shfl_xor(p, 2);
    p += __shfl_xor(p, 1);
    float h = fmaxf(p + b1[rr], 0.f);
    o0 += w2[rr] * h;
    o1 += w2[16 + rr] * h;
  }
  if (lane == 0) {
    dec[(size_t)t * 2]     = o0;
    dec[(size_t)t * 2 + 1] = o1;
  }
}

// ---------------------------------------------------------------------------
extern "C" void kernel_launch(void* const* d_in, const int* in_sizes, int n_in,
                              void* d_out, int out_size, void* d_ws, size_t ws_size,
                              hipStream_t stream) {
  (void)in_sizes; (void)n_in; (void)out_size; (void)ws_size;
  const float* traj  = (const float*)d_in[0];
  const float* ew1   = (const float*)d_in[1];
  const float* eb1   = (const float*)d_in[2];
  const float* ew2   = (const float*)d_in[3];
  const float* eb2   = (const float*)d_in[4];
  const float* trans = (const float*)d_in[5];
  const float* dw1   = (const float*)d_in[6];
  const float* db1   = (const float*)d_in[7];
  const float* dw2   = (const float*)d_in[8];
  const float* db2   = (const float*)d_in[9];

  char* ws = (char*)d_ws;
  _Float16*      ts16 = (_Float16*)(ws + WS_TS16);
  unsigned char* act  = (unsigned char*)(ws + WS_ACT);
  float*         s0   = (float*)(ws + WS_S0);
  float*         cst  = (float*)(ws + WS_CST);
  _Float16*      mt   = (_Float16*)(ws + WS_MT);

  float* preds = (float*)d_out;
  float* dec   = preds + (size_t)T_LEN * KDIM;

  softmax_T_kernel<<<512, 256, 0, stream>>>(trans, ts16);
  actions_kernel<<<T_LEN / 256, 256, 0, stream>>>(traj, act);
  s0_kernel<<<1, 256, 0, stream>>>(traj, ew1, eb1, ew2, eb2, s0);

  size_t lds_prod = (size_t)256 * PSTR * sizeof(_Float16);                 // ~132KB
  chunk_prod_kernel<<<NCHUNK, 1024, lds_prod, stream>>>(ts16, act, mt);

  scan_kernel<<<1, 256, 0, stream>>>(mt, s0, cst);

  size_t lds_rep = (size_t)2 * 256 * PSTR * sizeof(_Float16) + 512 * 4 + CLEN; // ~267KB
  replay_kernel<<<NCHUNK, 1024, lds_rep, stream>>>(ts16, act, cst, preds);

  decode_kernel<<<T_LEN / 8, 256, 0, stream>>>(preds, s0, dw1, db1, dw2, db2, dec);
}
